// TextConv2DLayer_22144851378518
// MI455X (gfx1250) — matrix operand order repaired and checked by reading, not compiled, not run on hardware
//
#include <hip/hip_runtime.h>

// ---------------------------------------------------------------------------
// Fused Conv(valid, K x 300) + LIF spiking scan for MI455X (gfx1250).
//
// Roofline: output = 393 MB of spikes -> store-bandwidth bound (~17us at
// 23.3 TB/s). Conv FLOPs (~2 GF) are trivial on the WMMA pipe; inputs fit
// in L2 (192 MB). Strategy: f16 WMMA GEMM (f32 accumulate) for the conv,
// LIF recurrence kept entirely in registers (the drive is time-constant),
// only the final spike stream goes to HBM.
//
// K-loop is unrolled 2x with two independent accumulators (padded K-dim is
// always a multiple of 64) so consecutive v_wmma ops have no D->C chain.
// ---------------------------------------------------------------------------

typedef __attribute__((ext_vector_type(16))) _Float16 v16h;
typedef __attribute__((ext_vector_type(8)))  _Float16 v8h;
typedef __attribute__((ext_vector_type(8)))  float    v8f;

#define BN 16    // batch
#define LT 128   // time steps
#define DI 300   // true embedding dim
#define DP 320   // padded dim (multiple of 32 for WMMA K-steps)
#define FN 128   // filters

// ---- prep: pad/convert x (B,L,300) f32 -> (B,L,320) f16, zeros in tail ----
__global__ void pad_x_kernel(const float* __restrict__ x, _Float16* __restrict__ xp) {
  int idx = blockIdx.x * blockDim.x + threadIdx.x;
  const int total = BN * LT * DP;
  if (idx >= total) return;
  int d   = idx % DP;
  int row = idx / DP;                       // b*L + l
  float v = (d < DI) ? x[row * DI + d] : 0.0f;
  xp[idx] = (_Float16)v;
}

// ---- prep: pad/convert W (F,1,K,300) f32 -> (F, K*320) f16 ----
__global__ void pad_w_kernel(const float* __restrict__ W, _Float16* __restrict__ Wp, int K) {
  int idx = blockIdx.x * blockDim.x + threadIdx.x;
  const int total = FN * K * DP;
  if (idx >= total) return;
  int d  = idx % DP;
  int fk = idx / DP;                        // f*K + k
  float v = (d < DI) ? W[fk * DI + d] : 0.0f;
  Wp[idx] = (_Float16)v;
}

// ---- fused WMMA conv + LIF scan; one wave per 16x16 (f,h) tile ----
template <int K>
__global__ __launch_bounds__(32) void lif_conv_wmma(
    const _Float16* __restrict__ xp,   // (B, L, 320) f16
    const _Float16* __restrict__ Wp,   // (F, K*320)  f16
    float* __restrict__ out) {         // (B, L, F, H) f32
  constexpr int K320 = K * DP;
  constexpr int H    = LT - K + 1;

  const int lane = threadIdx.x & 31;
  const int l16  = lane & 15;
  const int hi   = lane >> 4;           // half-wave selector per ISA layouts

  const int b  = blockIdx.z;
  const int f0 = blockIdx.y << 4;
  const int h0 = blockIdx.x << 4;

  const int h     = h0 + l16;
  const int hload = (h < H) ? h : (H - 1);   // clamp: keep EXEC all-1s for WMMA

  // A fragment source: row f = f0+l16 of Wp; lanes>=16 hold K-halves 8..15/24..31
  const _Float16* wrow = Wp + (size_t)(f0 + l16) * K320 + hi * 8;
  // B fragment source: column h of the implicit im2col; contiguous in d.
  // lanes>=16 hold K rows 8..15/24..31 of each 32-step -> +8 halves in d.
  const _Float16* xrow = xp + ((size_t)b * LT + hload) * DP + hi * 8;

  union Frag { v16h v; v8h p[2]; };
  Frag a0, a1, b0, b1;
  v8f acc0 = {}, acc1 = {};

  // K320 is a multiple of 64 for K in {3,4,5}; a 64-step never crosses a
  // tap boundary (DP=320, offsets within a tap stay < 320).
#pragma unroll 1
  for (int kk = 0; kk < K320; kk += 64) {
    const int ksl = kk / DP;            // conv tap index (same for both halves)
    const int d0  = kk % DP;            // offset within padded row

    const _Float16* pa = wrow + kk;
    const _Float16* pb = xrow + (size_t)ksl * DP + d0;

    a0.p[0] = *(const v8h*)(pa);
    a0.p[1] = *(const v8h*)(pa + 16);
    b0.p[0] = *(const v8h*)(pb);
    b0.p[1] = *(const v8h*)(pb + 16);

    a1.p[0] = *(const v8h*)(pa + 32);
    a1.p[1] = *(const v8h*)(pa + 48);
    b1.p[0] = *(const v8h*)(pb + 32);
    b1.p[1] = *(const v8h*)(pb + 48);

    // warm the next 64-step (gfx1250 global_prefetch_b8)
    __builtin_prefetch((const void*)(pa + 64), 0, 3);
    __builtin_prefetch((const void*)(pb + 64), 0, 3);

    // two independent accumulation chains -> back-to-back v_wmma without
    // a D->C hazard between them
    acc0 = __builtin_amdgcn_wmma_f32_16x16x32_f16(
        false, a0.v, false, b0.v, (short)0, acc0, false, false);
    acc1 = __builtin_amdgcn_wmma_f32_16x16x32_f16(
        false, a1.v, false, b1.v, (short)0, acc1, false, false);
  }

  v8f acc;
#pragma unroll
  for (int r = 0; r < 8; ++r) acc[r] = acc0[r] + acc1[r];

  // C/D layout: VGPR r -> f = f0 + r + 8*hi ; lane -> h = h0 + l16
  const int  fbase = f0 + hi * 8;
  const bool valid = (h < H);

  float vv[8] = {0.f, 0.f, 0.f, 0.f, 0.f, 0.f, 0.f, 0.f};
  float* outb = out + (size_t)b * LT * FN * H + (size_t)fbase * H + h;
  constexpr size_t tstride = (size_t)FN * H;

#pragma unroll 1
  for (int t = 0; t < LT; ++t) {
    float* o = outb + (size_t)t * tstride;
#pragma unroll
    for (int r = 0; r < 8; ++r) {
      vv[r] = 0.9f * vv[r] + acc[r];            // ALPHA = 0.9
      float s = (vv[r] >= 1.0f) ? 1.0f : 0.0f;  // VTH = 1.0 (forward = hard)
      vv[r] -= s;
      if (valid) o[(size_t)r * H] = s;
    }
  }
}

extern "C" void kernel_launch(void* const* d_in, const int* in_sizes, int n_in,
                              void* d_out, int out_size, void* d_ws, size_t ws_size,
                              hipStream_t stream) {
  (void)in_sizes; (void)n_in; (void)out_size; (void)ws_size;

  const float* x  = (const float*)d_in[0];   // (16,128,300)
  const float* W0 = (const float*)d_in[1];   // (128,1,3,300)
  const float* W1 = (const float*)d_in[2];   // (128,1,4,300)
  const float* W2 = (const float*)d_in[3];   // (128,1,5,300)

  // workspace layout (f16): xp | W0p | W1p | W2p  (~2.3 MB total)
  _Float16* xp  = (_Float16*)d_ws;
  const size_t xp_elems = (size_t)BN * LT * DP;          // 655,360
  _Float16* w0p = xp  + xp_elems;
  _Float16* w1p = w0p + (size_t)FN * 3 * DP;
  _Float16* w2p = w1p + (size_t)FN * 4 * DP;

  float* out0 = (float*)d_out;
  float* out1 = out0 + (size_t)BN * LT * FN * 126;
  float* out2 = out1 + (size_t)BN * LT * FN * 125;

  // prep: convert + pad to f16
  {
    int total = BN * LT * DP;
    pad_x_kernel<<<(total + 255) / 256, 256, 0, stream>>>(x, xp);
  }
  {
    int t0 = FN * 3 * DP, t1 = FN * 4 * DP, t2 = FN * 5 * DP;
    pad_w_kernel<<<(t0 + 255) / 256, 256, 0, stream>>>(W0, w0p, 3);
    pad_w_kernel<<<(t1 + 255) / 256, 256, 0, stream>>>(W1, w1p, 4);
    pad_w_kernel<<<(t2 + 255) / 256, 256, 0, stream>>>(W2, w2p, 5);
  }

  // fused WMMA conv + LIF per branch (H = 126 / 125 / 124, all tile to 8)
  {
    dim3 blk(32, 1, 1);
    dim3 g(8, FN / 16, BN);
    lif_conv_wmma<3><<<g, blk, 0, stream>>>(xp, w0p, out0);
    lif_conv_wmma<4><<<g, blk, 0, stream>>>(xp, w1p, out1);
    lif_conv_wmma<5><<<g, blk, 0, stream>>>(xp, w2p, out2);
  }
}
